// VFSMN_9861244911962
// MI455X (gfx1250) — matrix operand order repaired and checked by reading, not compiled, not run on hardware
//
#include <hip/hip_runtime.h>

// VFSMN fused kernel set for MI455X (gfx1250, wave32, WMMA bf16).
//   out = x@W1 + h@W2 + bias
//   h[t] = sum_{l=1..40} (mw[l]-mw[0]) * x[t-l]  +  mw[0]*total[b]   (per-channel)
// The mw[0]*total term is folded into a per-batch bias2 = bias + (mw0*total)@W2.
//
// Workspace layout (bytes), total 860,160:
//   [0)        partials  512*256 f32   (two-stage column sums, no atomics)
//   [524288)   cvec       16*256 f32   (mw0 * total)
//   [540672)   bias2      16*256 f32
//   [557056)   wprime     40*256 f32   (mw[l]-mw[0])
//   [598016)   W1p      65536 u16      (bf16, WMMA-B-fragment packed)
//   [729088)   W2p      65536 u16

#define BATCH 16
#define SEQ   4096
#define DIM   256
#define MEM   40
#define TM    64      // output rows per workgroup (main kernel)
#define HALO  (TM + MEM)   // 104 staged rows
#define LDB   264     // padded bf16 LDS row stride (256 + 8) -> conflict-free frag reads
#define TCH   32      // T-chunks for column-sum stage 1
#define ROWS_PER_CH (SEQ / TCH)   // 128

typedef __attribute__((ext_vector_type(4)))  float          f32x4;
typedef __attribute__((ext_vector_type(8)))  float          v8f;
typedef __attribute__((ext_vector_type(4)))  unsigned short u16x4;
typedef __attribute__((ext_vector_type(8)))  unsigned short u16x8;
typedef __attribute__((ext_vector_type(16))) unsigned short u16x16;
typedef __attribute__((ext_vector_type(16))) __bf16         v16bf;

union BFrag { u16x16 u; v16bf v; };

__device__ __forceinline__ unsigned short f2bf(float f) {
  unsigned u = __float_as_uint(f);
  u += 0x7fffu + ((u >> 16) & 1u);          // round-to-nearest-even
  return (unsigned short)(u >> 16);
}
__device__ __forceinline__ f32x4 bf4f(u16x4 h) {
  f32x4 r;
  r.x = __uint_as_float(((unsigned)h.x) << 16);
  r.y = __uint_as_float(((unsigned)h.y) << 16);
  r.z = __uint_as_float(((unsigned)h.z) << 16);
  r.w = __uint_as_float(((unsigned)h.w) << 16);
  return r;
}

// ---------- K1: partial column sums over T-chunks (deterministic, no atomics) ----------
__global__ __launch_bounds__(256) void k_partial(const float* __restrict__ x,
                                                 float* __restrict__ partials) {
  const int b = blockIdx.x >> 5;       // 16
  const int c = blockIdx.x & 31;       // 32 chunks
  const int d = threadIdx.x;
  const float* p = x + ((long)b * SEQ + (long)c * ROWS_PER_CH) * DIM + d;
  float s = 0.f;
  for (int t = 0; t < ROWS_PER_CH; ++t) s += p[t * DIM];
  partials[(b * TCH + c) * DIM + d] = s;
}

// ---------- K2: finalize totals, fold mw[0] -> cvec ----------
__global__ __launch_bounds__(256) void k_totals(const float* __restrict__ partials,
                                                const float* __restrict__ mw,
                                                float* __restrict__ cvec) {
  const int b = blockIdx.x, d = threadIdx.x;
  float s = 0.f;
  for (int c = 0; c < TCH; ++c) s += partials[(b * TCH + c) * DIM + d];
  cvec[b * DIM + d] = s * mw[d];   // mw row 0
}

// ---------- K3: bias2[b,n] = bias[n] + cvec[b,:] @ W2 ----------
__global__ __launch_bounds__(256) void k_bias2(const float* __restrict__ cvec,
                                               const float* __restrict__ W2,
                                               const float* __restrict__ bias,
                                               float* __restrict__ bias2) {
  const int b = blockIdx.x, n = threadIdx.x;
  float s = bias[n];
  const float* cv = cvec + b * DIM;
  for (int d = 0; d < DIM; ++d) s += cv[d] * W2[d * DIM + n];
  bias2[b * DIM + n] = s;
}

// ---------- K4: wprime[l-1,d] = mw[l,d] - mw[0,d] ----------
__global__ __launch_bounds__(256) void k_wprime(const float* __restrict__ mw,
                                                float* __restrict__ wp) {
  const int l = blockIdx.x, d = threadIdx.x;
  wp[l * DIM + d] = mw[(l + 1) * DIM + d] - mw[d];
}

// ---------- K5: pack W1/W2 (fp32 row-major) into bf16 WMMA B-fragment order ----------
// Fragment (ks,nb): 32 lanes x 16 bf16, lane-contiguous.
// B 32x16 bf16 layout: lane L -> n = nb*16 + (L&15); k = ks*32 + ((e&8)?16:0) + ((L&16)?8:0) + (e&7).
__global__ __launch_bounds__(256) void k_pack(const float* __restrict__ W1,
                                              const float* __restrict__ W2,
                                              unsigned short* __restrict__ W1p,
                                              unsigned short* __restrict__ W2p) {
  const int p    = blockIdx.x * 256 + threadIdx.x;   // 0..65535
  const int e    = p & 15;
  const int lane = (p >> 4) & 31;
  const int nb   = (p >> 9) & 15;
  const int ks   = p >> 13;
  const int k = ks * 32 + ((e & 8) ? 16 : 0) + ((lane & 16) ? 8 : 0) + (e & 7);
  const int n = nb * 16 + (lane & 15);
  W1p[p] = f2bf(W1[k * DIM + n]);
  W2p[p] = f2bf(W2[k * DIM + n]);
}

// ---------- K6: fused conv + dual bf16 WMMA GEMM ----------
__global__ __launch_bounds__(256) void k_main(const float* __restrict__ x,
                                              const unsigned short* __restrict__ W1p,
                                              const unsigned short* __restrict__ W2p,
                                              const float* __restrict__ wp,
                                              const float* __restrict__ bias2,
                                              float* __restrict__ out) {
  __shared__ unsigned short Xsb[HALO * LDB];  // bf16 x rows [t0-40, t0+63]
  __shared__ unsigned short Hsb[TM * LDB];    // bf16 conv output rows [t0, t0+63]

  const int tid = threadIdx.x;
  const int b   = blockIdx.x >> 6;            // SEQ/TM = 64 chunks per batch
  const int t0  = (blockIdx.x & 63) * TM;

  // --- stage x -> bf16 LDS (zero-fill causal halo before t=0) ---
  {
    const int rr = tid >> 6;                  // 4 rows per pass
    const int q  = tid & 63;                  // float4 column
    for (int rbase = 0; rbase < HALO; rbase += 4) {
      const int r = rbase + rr;
      const int t = t0 - MEM + r;
      f32x4 v = {0.f, 0.f, 0.f, 0.f};
      if (t >= 0) v = *(const f32x4*)&x[(((long)b * SEQ + t) * DIM) + q * 4];
      u16x4 h;
      h.x = f2bf(v.x); h.y = f2bf(v.y); h.z = f2bf(v.z); h.w = f2bf(v.w);
      *(u16x4*)&Xsb[r * LDB + q * 4] = h;
    }
  }
  __syncthreads();

  // --- 40-tap depthwise conv: H[i,d] = sum_l wp[l-1,d] * X[i-l,d] ---
  {
    const int q  = tid & 63;                  // d-group of 4
    const int d  = q * 4;
    const int ig = tid >> 6;                  // 16 rows each
    f32x4 acc[16];
#pragma unroll
    for (int i = 0; i < 16; ++i) acc[i] = (f32x4){0.f, 0.f, 0.f, 0.f};
    for (int l = 1; l <= MEM; ++l) {
      const f32x4 w = *(const f32x4*)&wp[(l - 1) * DIM + d];
#pragma unroll
      for (int i = 0; i < 16; ++i) {
        const int row = MEM + ig * 16 + i - l;
        acc[i] += w * bf4f(*(const u16x4*)&Xsb[row * LDB + d]);
      }
    }
#pragma unroll
    for (int i = 0; i < 16; ++i) {
      u16x4 h;
      h.x = f2bf(acc[i].x); h.y = f2bf(acc[i].y);
      h.z = f2bf(acc[i].z); h.w = f2bf(acc[i].w);
      *(u16x4*)&Hsb[(ig * 16 + i) * LDB + d] = h;
    }
  }
  __syncthreads();

  // --- GEMM: out_tile = X@W1 + H@W2 (+bias2) via v_wmma_f32_16x16x32_bf16 ---
  const int lane = tid & 31;
  const int wave = tid >> 5;       // 8 waves
  const int mb   = wave >> 1;      // 4 row blocks of 16
  const int nh   = wave & 1;       // 2 halves of 8 n-blocks
  const int nlo  = lane & 15;
  const int kofsA = (lane >> 4) * 8;   // A layout: upper lane-half holds K+8

  v8f acc[8];
#pragma unroll
  for (int j = 0; j < 8; ++j) acc[j] = (v8f){0.f,0.f,0.f,0.f,0.f,0.f,0.f,0.f};

#pragma unroll
  for (int ks = 0; ks < 8; ++ks) {
    // A fragments (16x32 bf16): two contiguous 16B runs at k = base and base+16
    const int e0 = ks * 32 + kofsA;
    const u16x8* px = (const u16x8*)&Xsb[(MEM + mb * 16 + nlo) * LDB + e0];
    const u16x8* ph = (const u16x8*)&Hsb[(mb * 16 + nlo) * LDB + e0];
    BFrag aX, aH;
    aX.u = __builtin_shufflevector(px[0], px[2], 0,1,2,3,4,5,6,7,8,9,10,11,12,13,14,15);
    aH.u = __builtin_shufflevector(ph[0], ph[2], 0,1,2,3,4,5,6,7,8,9,10,11,12,13,14,15);
#pragma unroll
    for (int j = 0; j < 8; ++j) {
      const int nb = nh * 8 + j;
      const u16x8* p1 = (const u16x8*)(W1p + (((ks * 16 + nb) * 32) + lane) * 16);
      const u16x8* p2 = (const u16x8*)(W2p + (((ks * 16 + nb) * 32) + lane) * 16);
      BFrag b1, b2;
      b1.u = __builtin_shufflevector(p1[0], p1[1], 0,1,2,3,4,5,6,7,8,9,10,11,12,13,14,15);
      b2.u = __builtin_shufflevector(p2[0], p2[1], 0,1,2,3,4,5,6,7,8,9,10,11,12,13,14,15);
      acc[j] = __builtin_amdgcn_wmma_f32_16x16x32_bf16(false, aX.v, false, b1.v,
                                                       (short)0, acc[j], false, false);
      acc[j] = __builtin_amdgcn_wmma_f32_16x16x32_bf16(false, aH.v, false, b2.v,
                                                       (short)0, acc[j], false, false);
    }
  }

  // --- epilogue: C/D layout lanes0-15 M=r, lanes16-31 M=r+8; add bias2, store fp32 ---
  const float* b2row = bias2 + b * DIM;
  const int mhalf = (lane >> 4) * 8;
#pragma unroll
  for (int j = 0; j < 8; ++j) {
    const int n  = (nh * 8 + j) * 16 + nlo;
    const float bb = b2row[n];
#pragma unroll
    for (int r = 0; r < 8; ++r) {
      const int t = t0 + mb * 16 + mhalf + r;
      out[(((long)b * SEQ + t) * DIM) + n] = acc[j][r] + bb;
    }
  }
}

extern "C" void kernel_launch(void* const* d_in, const int* in_sizes, int n_in,
                              void* d_out, int out_size, void* d_ws, size_t ws_size,
                              hipStream_t stream) {
  const float* x    = (const float*)d_in[0];
  const float* W1   = (const float*)d_in[1];
  const float* W2   = (const float*)d_in[2];
  const float* bias = (const float*)d_in[3];
  const float* mw   = (const float*)d_in[4];
  float* out = (float*)d_out;

  char* ws = (char*)d_ws;
  float*          partials = (float*)(ws);
  float*          cvec     = (float*)(ws + 524288);
  float*          bias2    = (float*)(ws + 540672);
  float*          wp       = (float*)(ws + 557056);
  unsigned short* W1p      = (unsigned short*)(ws + 598016);
  unsigned short* W2p      = (unsigned short*)(ws + 729088);

  k_partial<<<dim3(BATCH * TCH), dim3(256), 0, stream>>>(x, partials);
  k_totals <<<dim3(BATCH),       dim3(256), 0, stream>>>(partials, mw, cvec);
  k_bias2  <<<dim3(BATCH),       dim3(256), 0, stream>>>(cvec, W2, bias, bias2);
  k_wprime <<<dim3(MEM),         dim3(256), 0, stream>>>(mw, wp);
  k_pack   <<<dim3(256),         dim3(256), 0, stream>>>(W1, W2, W1p, W2p);
  k_main   <<<dim3(BATCH * (SEQ / TM)), dim3(256), 0, stream>>>(x, W1p, W2p, wp, bias2, out);
}
